// MyModel_25572235280571
// MI455X (gfx1250) — matrix operand order repaired and checked by reading, compiled, not run on hardware
//
#include <hip/hip_runtime.h>
#include <hip/hip_bf16.h>
#include <math.h>

typedef _Float16 f16;
typedef f16 v16h __attribute__((ext_vector_type(16)));
typedef f16 v8h  __attribute__((ext_vector_type(8)));
typedef f16 v4h  __attribute__((ext_vector_type(4)));
typedef float v8f __attribute__((ext_vector_type(8)));

#define B_  256
#define T_  200
#define E_  100
#define H_  128
#define K_  13
#define G4  512            // 4*H
#define BT  (B_*T_)        // 51200
#define EP  128            // E padded to multiple of 32

// ---- workspace offsets (bytes), 16B aligned ----
#define OFF_XF16  ((size_t)0)            // BT*EP*2        = 13107200
#define OFF_WT    ((size_t)13107200)     // 2*512*128*2    = 262144   (W transposed, f16)
#define OFF_UT    ((size_t)13369344)     // 2*512*128*2    = 262144   (U transposed, f16)
#define OFF_XW    ((size_t)13631488)     // 2*BT*512*2     = 104857600 (gate pre-acts, f16,
                                         //   layout [dir][t][b][hcol][gate])
#define OFF_HALL  ((size_t)118489088)    // BT*256*2       = 26214400 (hidden states, f16)
#define OFF_LENS  ((size_t)144703488)    // B_*4

static __device__ __forceinline__ v8f wmma16(v16h a, v16h b, v8f c) {
  return __builtin_amdgcn_wmma_f32_16x16x32_f16(false, a, false, b, (short)0, c,
                                                false, false);
}

// A fragment: 16x32 tile, source row-major [rows][ld] f16.
// lanes 0-15: M=lane, K {k0+0..7, k0+16..23}; lanes 16-31: M=lane-16, K {+8.., +24..}
static __device__ __forceinline__ v16h load_a(const f16* base, int row0, int k0,
                                              int ld, int lane) {
  int r  = row0 + (lane & 15);
  int ko = k0 + ((lane >> 4) << 3);
  const f16* p = base + (size_t)r * ld + ko;
  v8h lo = *(const v8h*)(p);
  v8h hi = *(const v8h*)(p + 16);
  return __builtin_shufflevector(lo, hi, 0,1,2,3,4,5,6,7,8,9,10,11,12,13,14,15);
}

// B fragment: 32x16 tile from K-major storage src[n][k] ([N][ld] f16).
// lanes 0-15: N=lane, K {k0+0..7, k0+16..23}; lanes 16-31: N=lane-16, K {+8.., +24..}
static __device__ __forceinline__ v16h load_b(const f16* base, int n0, int k0,
                                              int ld, int lane) {
  int n  = n0 + (lane & 15);
  int ko = k0 + ((lane >> 4) << 3);
  const f16* p = base + (size_t)n * ld + ko;
  v8h lo = *(const v8h*)(p);
  v8h hi = *(const v8h*)(p + 16);
  return __builtin_shufflevector(lo, hi, 0,1,2,3,4,5,6,7,8,9,10,11,12,13,14,15);
}

// ---------------- kernel 1: sequence lengths ----------------
__global__ void k_lens(const int* __restrict__ text, int* __restrict__ lens_ws,
                       float* __restrict__ lens_out) {
  int b = threadIdx.x;
  int c = 0;
  for (int t = 0; t < T_; ++t) c += (text[b * T_ + t] != 0) ? 1 : 0;
  lens_ws[b] = c;
  lens_out[b] = (float)c;
}

// ---------------- kernel 2: embedding gather -> f16, pad E 100->128 ----------------
__global__ void k_embed(const int* __restrict__ text, const float* __restrict__ emb,
                        f16* __restrict__ xf) {
  int p = blockIdx.x * 256 + threadIdx.x;        // 0..BT-1
  int tok = text[p];
  const float4* e4 = (const float4*)(emb + (size_t)tok * E_);   // 400B rows, 16B aligned
  f16* o = xf + (size_t)p * EP;
#pragma unroll
  for (int i = 0; i < 25; ++i) {
    float4 v = e4[i];
    o[i * 4 + 0] = (f16)v.x;
    o[i * 4 + 1] = (f16)v.y;
    o[i * 4 + 2] = (f16)v.z;
    o[i * 4 + 3] = (f16)v.w;
  }
#pragma unroll
  for (int i = E_; i < EP; ++i) o[i] = (f16)0.0f;
}

// ---------------- kernel 3: transpose+convert W/U to K-major f16 ----------------
__global__ void k_convert(const float* __restrict__ Wf, const float* __restrict__ Uf,
                          const float* __restrict__ Wb, const float* __restrict__ Ub,
                          f16* __restrict__ Wt, f16* __restrict__ Ut) {
  int idx = blockIdx.x * 256 + threadIdx.x;      // 0..262143
  int a   = idx >> 16;                           // 0..3 : Wf, Wb, Uf, Ub
  int rem = idx & 65535;
  int k   = rem >> 9;                            // 0..127
  int n   = rem & 511;                           // 0..511
  if (a < 2) {
    const float* src = (a == 0) ? Wf : Wb;       // [E_=100][512]
    f16* dst = Wt + (size_t)a * 65536;           // [512][128]
    dst[(size_t)n * 128 + k] = (k < E_) ? (f16)src[(size_t)k * 512 + n] : (f16)0.0f;
  } else {
    const float* src = (a == 2) ? Uf : Ub;       // [128][512]
    f16* dst = Ut + (size_t)(a - 2) * 65536;     // [512][128]
    dst[(size_t)n * 128 + k] = (f16)src[(size_t)k * 512 + n];
  }
}

// ---------------- kernel 4: input projection xw = x @ W + b (both dirs) ----------------
// output layout: xw[dir][t][b][hcol][gate]  (gate innermost -> scan loads v4h per (b,hcol))
__global__ __launch_bounds__(256) void k_inproj(const f16* __restrict__ xf,
                                                const f16* __restrict__ Wt,
                                                const float* __restrict__ bf,
                                                const float* __restrict__ bb,
                                                f16* __restrict__ xw) {
  int wid  = blockIdx.x * 8 + (threadIdx.x >> 5);
  int lane = threadIdx.x & 31;
  int mt = wid >> 6;                 // 0..3199 row tile
  int ct = wid & 63;                 // dir*32 + nt
  int dir = ct >> 5, nt = ct & 31;
  const f16* W = Wt + (size_t)dir * 65536;
  const float* bias = dir ? bb : bf;

  v8f acc;
#pragma unroll
  for (int i = 0; i < 8; ++i) acc[i] = 0.0f;
#pragma unroll
  for (int k4 = 0; k4 < 4; ++k4) {
    v16h A = load_a(xf, mt * 16, k4 * 32, EP, lane);
    v16h Bf = load_b(W, nt * 16, k4 * 32, 128, lane);
    acc = wmma16(A, Bf, acc);
  }
  int col = nt * 16 + (lane & 15);   // 0..511 = gate*128 + hcol
  int g   = col >> 7;
  int nl  = col & 127;
  float bv = bias[col];
  int rbase = mt * 16 + ((lane >> 4) << 3);
  // rbase is a multiple of 8 and 200 % 8 == 0, so t does not wrap within the 8 rows
  int t0 = rbase % T_;
  int b0 = rbase / T_;
  f16* out = xw + (size_t)dir * BT * G4;
#pragma unroll
  for (int i = 0; i < 8; ++i) {
    size_t o = (((size_t)(t0 + i) * B_ + b0) * H_ + nl) * 4 + g;
    out[o] = (f16)(acc[i] + bv);
  }
}

// ---------------- kernel 5: persistent LSTM scan (1 block per direction) ----------------
// LDS: U^T (512x128 f16 = 128KB) + h (256x128 f16 = 64KB) = 192KB of the WGP's 320KB.
// c state lives in VGPRs. 32 waves; wave w owns M-tile w/2 and 4 H-column tiles.
__global__ __launch_bounds__(1024) void k_scan(const f16* __restrict__ Ut_all,
                                               const f16* __restrict__ xw_all,
                                               f16* __restrict__ h_all) {
  extern __shared__ char smem[];
  f16* sU = (f16*)smem;                          // [512][128]
  f16* sH = (f16*)(smem + 512 * 128 * 2);        // [256][128]

  const int dir = blockIdx.x;
  const int tid = threadIdx.x;
  const int wave = tid >> 5, lane = tid & 31;
  const int mt  = wave >> 1;            // 0..15 (batch rows mt*16..)
  const int nh0 = (wave & 1) * 4;       // h-column tile base (each tile = 16 cols)

  // stage U^T into LDS, zero h
  const f16* Usrc = Ut_all + (size_t)dir * 65536;
  for (int i = tid; i < 512 * 128 / 8; i += 1024)
    ((v8h*)sU)[i] = ((const v8h*)Usrc)[i];
  v8h z;
#pragma unroll
  for (int i = 0; i < 8; ++i) z[i] = (f16)0.0f;
  for (int i = tid; i < 256 * 128 / 8; i += 1024) ((v8h*)sH)[i] = z;

  v8f creg[4];
#pragma unroll
  for (int g = 0; g < 4; ++g)
#pragma unroll
    for (int i = 0; i < 8; ++i) creg[g][i] = 0.0f;

  const f16* xw = xw_all + (size_t)dir * BT * G4;   // [t][b][hcol][gate]
  const int rbase = mt * 16 + ((lane >> 4) << 3);

  __syncthreads();

  for (int s = 0; s < T_; ++s) {
    const int t = dir ? (T_ - 1 - s) : s;

    // prefetch next timestep's xw slice for this thread's rows (global_prefetch_b8)
    {
      int tn = dir ? (t - 1) : (t + 1);
      if (tn < 0) tn = 0;
      if (tn >= T_) tn = T_ - 1;
#pragma unroll
      for (int g4 = 0; g4 < 4; ++g4) {
        const f16* pf = xw + (((size_t)tn * B_ + rbase) * H_ +
                              (nh0 + g4) * 16 + (lane & 15)) * 4;
        __builtin_prefetch((const void*)pf, 0, 3);
      }
    }

    // A fragments of current h (shared across this wave's 16 tiles)
    v16h A[4];
#pragma unroll
    for (int k4 = 0; k4 < 4; ++k4) A[k4] = load_a(sH, mt * 16, k4 * 32, 128, lane);
    __syncthreads();   // everyone captured old h before anyone overwrites it

#pragma unroll
    for (int g4 = 0; g4 < 4; ++g4) {
      const int nh = nh0 + g4;
      const int nloc = nh * 16 + (lane & 15);    // h column 0..127
      v8f acc[4];
#pragma unroll
      for (int g = 0; g < 4; ++g)
#pragma unroll
        for (int i = 0; i < 8; ++i) acc[g][i] = 0.0f;

#pragma unroll
      for (int k4 = 0; k4 < 4; ++k4) {
#pragma unroll
        for (int g = 0; g < 4; ++g) {
          v16h Bf = load_b(sU, g * H_ + nh * 16, k4 * 32, 128, lane);
          acc[g] = wmma16(A[k4], Bf, acc[g]);
        }
      }

#pragma unroll
      for (int i = 0; i < 8; ++i) {
        const int b = rbase + i;
        // one aligned 8B load gives all 4 gate pre-activations for (b, nloc)
        v4h xg = *(const v4h*)(xw + (((size_t)t * B_ + b) * H_ + nloc) * 4);
        float zi = acc[0][i] + (float)xg[0];
        float zf = acc[1][i] + (float)xg[1];
        float zg = acc[2][i] + (float)xg[2];
        float zo = acc[3][i] + (float)xg[3];
        float ig = 1.0f / (1.0f + __expf(-zi));
        float fg = 1.0f / (1.0f + __expf(-zf));
        float gg = fmaxf(zg, 0.0f);
        float og = 1.0f / (1.0f + __expf(-zo));
        float c = fg * creg[g4][i] + ig * gg;
        creg[g4][i] = c;
        float h = og * fmaxf(c, 0.0f);
        f16 hh = (f16)h;
        sH[(size_t)b * 128 + nloc] = hh;
        h_all[((size_t)b * T_ + t) * 256 + dir * H_ + nloc] = hh;
      }
    }
    __syncthreads();   // new h visible before next step's A loads
  }
}

// ---------------- kernel 6: dense (2H -> K=13) + softmax ----------------
__global__ __launch_bounds__(256) void k_dense(const f16* __restrict__ h_all,
                                               const float* __restrict__ Wd,
                                               const float* __restrict__ bd,
                                               float* __restrict__ out) {
  __shared__ float sW[256 * K_];
  __shared__ float sb[K_];
  int tid = threadIdx.x;
  for (int i = tid; i < 256 * K_; i += 256) sW[i] = Wd[i];
  if (tid < K_) sb[tid] = bd[tid];
  __syncthreads();

  int p = blockIdx.x * 256 + tid;                // 0..BT-1
  float acc[K_];
#pragma unroll
  for (int k = 0; k < K_; ++k) acc[k] = sb[k];
  const f16* hp = h_all + (size_t)p * 256;
  for (int j0 = 0; j0 < 256; j0 += 8) {
    v8h hv = *(const v8h*)(hp + j0);             // 512B-aligned base
#pragma unroll
    for (int u = 0; u < 8; ++u) {
      float h = (float)hv[u];
#pragma unroll
      for (int k = 0; k < K_; ++k) acc[k] += h * sW[(j0 + u) * K_ + k];
    }
  }
  float m = acc[0];
#pragma unroll
  for (int k = 1; k < K_; ++k) m = fmaxf(m, acc[k]);
  float ssum = 0.0f;
#pragma unroll
  for (int k = 0; k < K_; ++k) { acc[k] = __expf(acc[k] - m); ssum += acc[k]; }
  float inv = 1.0f / ssum;
#pragma unroll
  for (int k = 0; k < K_; ++k) out[(size_t)p * K_ + k] = acc[k] * inv;
}

// ---------------- kernel 7: CRF log-likelihood (one wave per sequence) ----------------
__global__ __launch_bounds__(256) void k_crf(const float* __restrict__ logits,
                                             const int* __restrict__ labels,
                                             const int* __restrict__ lens,
                                             const float* __restrict__ trans,
                                             float* __restrict__ ll_out) {
  __shared__ float sT[K_ * K_];
  int tid = threadIdx.x;
  for (int i = tid; i < K_ * K_; i += 256) sT[i] = trans[i];
  __syncthreads();

  int wave = tid >> 5, lane = tid & 31;
  int b = blockIdx.x * 8 + wave;
  int len = lens[b];
  const float* lg = logits + (size_t)b * T_ * K_;
  const int* lab = labels + (size_t)b * T_;

  // unary + pairwise scores (lanes strided over t, wave-reduced)
  float us = 0.0f, bs = 0.0f;
  for (int t = lane; t < T_; t += 32) {
    if (t < len) {
      us += lg[t * K_ + lab[t]];
      if (t >= 1) bs += sT[lab[t - 1] * K_ + lab[t]];
    }
  }
  for (int off = 16; off > 0; off >>= 1) {
    us += __shfl_down(us, off);
    bs += __shfl_down(bs, off);
  }
  us = __shfl(us, 0);
  bs = __shfl(bs, 0);

  // forward algorithm: alpha state j in lane j (j<13)
  int j = (lane < K_) ? lane : (K_ - 1);
  float alpha = lg[0 * K_ + j];
  for (int t = 1; t < T_; ++t) {
    if (t < len) {
      float m = -__builtin_inff(), ssum = 0.0f;
#pragma unroll
      for (int i = 0; i < K_; ++i) {
        float ai = __shfl(alpha, i);
        float v = ai + sT[i * K_ + j];
        if (v > m) { ssum = ssum * __expf(m - v) + 1.0f; m = v; }
        else       { ssum += __expf(v - m); }
      }
      alpha = m + __logf(ssum) + lg[t * K_ + j];
    }
  }
  float av = (lane < K_) ? alpha : -__builtin_inff();
  float m = av;
  for (int off = 16; off > 0; off >>= 1) m = fmaxf(m, __shfl_down(m, off));
  m = __shfl(m, 0);
  float e = (lane < K_) ? __expf(av - m) : 0.0f;
  for (int off = 16; off > 0; off >>= 1) e += __shfl_down(e, off);
  if (lane == 0) ll_out[b] = us + bs - (m + __logf(e));
}

// ---------------- host ----------------
extern "C" void kernel_launch(void* const* d_in, const int* in_sizes, int n_in,
                              void* d_out, int out_size, void* d_ws, size_t ws_size,
                              hipStream_t stream) {
  (void)in_sizes; (void)n_in; (void)out_size; (void)ws_size;
  const int*   text  = (const int*)d_in[0];
  const int*   lab   = (const int*)d_in[1];
  const float* emb   = (const float*)d_in[2];
  const float* Wf    = (const float*)d_in[3];
  const float* Uf    = (const float*)d_in[4];
  const float* bf    = (const float*)d_in[5];
  const float* Wb    = (const float*)d_in[6];
  const float* Ub    = (const float*)d_in[7];
  const float* bb    = (const float*)d_in[8];
  const float* Wd    = (const float*)d_in[9];
  const float* bd    = (const float*)d_in[10];
  const float* trans = (const float*)d_in[11];

  char* ws = (char*)d_ws;
  f16*  xf16 = (f16*)(ws + OFF_XF16);
  f16*  Wt   = (f16*)(ws + OFF_WT);
  f16*  Ut   = (f16*)(ws + OFF_UT);
  f16*  xw   = (f16*)(ws + OFF_XW);
  f16*  hall = (f16*)(ws + OFF_HALL);
  int*  lens = (int*)(ws + OFF_LENS);

  float* out    = (float*)d_out;
  float* out_lg = out;                         // [B,T,K] = 665600
  float* out_ln = out + (size_t)BT * K_;       // [B]     at 665600
  float* out_ll = out_ln + B_;                 // [B]     at 665856

  k_lens<<<1, 256, 0, stream>>>(text, lens, out_ln);
  k_embed<<<BT / 256, 256, 0, stream>>>(text, emb, xf16);
  k_convert<<<1024, 256, 0, stream>>>(Wf, Uf, Wb, Ub, Wt, Ut);
  k_inproj<<<25600, 256, 0, stream>>>(xf16, Wt, bf, bb, xw);
  k_scan<<<2, 1024, (512 * 128 + 256 * 128) * 2, stream>>>(Ut, xw, hall);
  k_dense<<<BT / 256, 256, 0, stream>>>(hall, Wd, bd, out_lg);
  k_crf<<<B_ / 8, 256, 0, stream>>>(out_lg, lab, lens, trans, out_ll);
}
